// Dti_cnn_mutiout_60473139527725
// MI455X (gfx1250) — compile-verified, matrix operand order checked by reading, not run on hardware
//
#include <hip/hip_runtime.h>
#include <hip/hip_bf16.h>
#include <math.h>

// ---------------------------------------------------------------------------
// Problem dims (compile-time constants from the reference)
// ---------------------------------------------------------------------------
#define DIMC 512
#define QKD  128
#define HIDC 1024            // HID
#define BB   8
#define SS   2048
#define BSR  (BB * SS)       // 16384 flattened rows

typedef __attribute__((ext_vector_type(16))) _Float16 v16h;
typedef __attribute__((ext_vector_type(8)))  _Float16 v8h;
typedef __attribute__((ext_vector_type(8)))  float    v8f;
typedef __attribute__((ext_vector_type(4)))  unsigned int u32x4;
typedef __attribute__((ext_vector_type(8)))  unsigned int u32x8;

// ---------------------------------------------------------------------------
// WMMA helper (CDNA5: V_WMMA_F32_16X16X32_F16, wave32)
// ---------------------------------------------------------------------------
__device__ __forceinline__ v8f wmma_f16(v16h a, v16h b, v8f c) {
  // 8 args: (neg_a, A, neg_b, B, c_mod, C, reuse_a, reuse_b)
  return __builtin_amdgcn_wmma_f32_16x16x32_f16(false, a, false, b, (short)0, c,
                                                false, false);
}

// Load a 16x32 fragment from a row-major f16 matrix (leading dim `ld` elems).
// A operand -> rows are M; B operand -> pass B^T (rows are N, cols are K).
// ISA 16-bit A layout: lane L: row=L&15, half=L>>4;
//   e<8 : k = e + 8*half; e>=8: k = 16 + (e-8) + 8*half  (two 16B chunks)
__device__ __forceinline__ v16h load_frag(const _Float16* __restrict__ p, int ld) {
  const int lane = threadIdx.x & 31;
  const int r    = lane & 15;
  const int h    = (lane >> 4) << 3;       // 0 or 8
  const _Float16* row = p + (size_t)r * ld + h;
  v8h lo = *(const v8h*)(row);
  v8h hi = *(const v8h*)(row + 16);
  v16h f;
#pragma unroll
  for (int e = 0; e < 8; ++e) { f[e] = lo[e]; f[8 + e] = hi[e]; }
  return f;
}

__device__ __forceinline__ float silu_f(float x) {
  return x / (1.0f + __expf(-x));
}

// ---------------------------------------------------------------------------
// Tensor Data Mover: DMA a 2D tile (rows x row_u64 8-byte elems, contiguous)
// from global memory into LDS. D# per CDNA5 ISA.
// Issued by one wave; caller waits TENSORcnt then barriers.
// ---------------------------------------------------------------------------
__device__ __forceinline__ void tdm_load_2d(unsigned int lds_off,
                                            const void* gptr,
                                            unsigned int row_u64,
                                            unsigned int rows) {
  const unsigned long long ga = (unsigned long long)(uintptr_t)gptr;
  u32x4 g0;
  g0[0] = 1u;                                        // count=1, user mode
  g0[1] = lds_off;                                   // lds_addr (bytes)
  g0[2] = (unsigned int)(ga & 0xffffffffu);          // global_addr[31:0]
  g0[3] = (unsigned int)((ga >> 32) & 0x1ffffffu)    // global_addr[56:32]
          | (2u << 30);                              // type = 2 ("image")
  u32x8 g1;
  g1[0] = (3u << 16);                                // data_size=3 (8B), no flags
  g1[1] = (row_u64 & 0xffffu) << 16;                 // tensor_dim0[15:0]
  g1[2] = (row_u64 >> 16) | ((rows & 0xffffu) << 16);// td0 hi | tensor_dim1 lo
  g1[3] = (rows >> 16) | ((row_u64 & 0xffffu) << 16);// td1 hi | tile_dim0
  g1[4] = rows & 0xffffu;                            // tile_dim1 | tile_dim2=0
  g1[5] = row_u64;                                   // tensor_dim0_stride[31:0]
  g1[6] = 0u;                                        // stride0 hi | stride1 lo
  g1[7] = 0u;                                        // stride1 hi
  const u32x4 gz = {0u, 0u, 0u, 0u};                 // groups 2/3: dims unused
  asm volatile("tensor_load_to_lds %0, %1, %2, %3"
               :: "s"(g0), "s"(g1), "s"(gz), "s"(gz)
               : "memory");
}

// ---------------------------------------------------------------------------
// K0: LayerNorm  x(f32, BSRxDIM) -> nx(f16).  One wave32 per row.
// ---------------------------------------------------------------------------
__global__ __launch_bounds__(256) void k_layernorm(
    const float* __restrict__ x, const float* __restrict__ g,
    const float* __restrict__ b, _Float16* __restrict__ nx) {
  const int wave = (blockIdx.x * blockDim.x + threadIdx.x) >> 5;
  const int lane = threadIdx.x & 31;
  const float* row = x + (size_t)wave * DIMC;
  float vals[16];
  float s = 0.f, s2 = 0.f;
#pragma unroll
  for (int i = 0; i < 16; ++i) {
    float v = row[lane + 32 * i];
    vals[i] = v; s += v; s2 += v * v;
  }
#pragma unroll
  for (int m = 16; m >= 1; m >>= 1) {
    s  += __shfl_xor(s,  m, 32);
    s2 += __shfl_xor(s2, m, 32);
  }
  const float mu  = s * (1.0f / DIMC);
  const float var = s2 * (1.0f / DIMC) - mu * mu;
  const float inv = rsqrtf(var + 1e-5f);
  _Float16* orow = nx + (size_t)wave * DIMC;
#pragma unroll
  for (int i = 0; i < 16; ++i) {
    const int c = lane + 32 * i;
    orow[c] = (_Float16)((vals[i] - mu) * inv * g[c] + b[c]);
  }
}

// ---------------------------------------------------------------------------
// K_prep: transpose f32 (K x N) -> f16 (N x K)  (weights -> B^T layout)
// ---------------------------------------------------------------------------
__global__ __launch_bounds__(256) void k_transpose_w(
    const float* __restrict__ src, _Float16* __restrict__ dst, int K, int N) {
  const int idx = blockIdx.x * blockDim.x + threadIdx.x;
  if (idx >= K * N) return;
  const int k = idx / N, n = idx % N;
  dst[(size_t)n * K + k] = (_Float16)src[idx];
}

// ---------------------------------------------------------------------------
// K1: h = silu(nx @ W_hidden + b)   (BSR x 2048, K=512)
//     cols [0,1024)   -> vT[b][d][s]  (f16, d-major)
//     cols [1024,2048)-> gate[s][d]   (f16)
// WG tile 128x64, 8 waves (4m x 2n). A-panel (128x512 f16 = 128 KB) staged in
// LDS by the Tensor Data Mover; B streamed from global.
// ---------------------------------------------------------------------------
__global__ __launch_bounds__(256) void k_gemm_hidden(
    const _Float16* __restrict__ nx, const _Float16* __restrict__ whT,
    const float* __restrict__ bh, _Float16* __restrict__ vT,
    _Float16* __restrict__ gate) {
  extern __shared__ __align__(16) _Float16 lA[];     // 128 x 512 f16
  const int wave = threadIdx.x >> 5;
  const int wm = wave >> 1, wn = wave & 1;
  const int mblk = blockIdx.x * 128;
  const int m0l = wm * 32;                           // local row in LDS panel
  const int n0 = blockIdx.y * 64 + wn * 32;

  if (threadIdx.x < 32) {
    tdm_load_2d((unsigned int)(uintptr_t)lA, nx + (size_t)mblk * DIMC,
                DIMC * 2 / 8, 128);
    __builtin_amdgcn_s_wait_tensorcnt(0);
  }
  __syncthreads();

  v8f acc[2][2] = {};
  for (int k = 0; k < DIMC; k += 32) {
    __builtin_prefetch(whT + (size_t)n0 * DIMC + k + 128, 0, 1);
    v16h a0 = load_frag(lA + (size_t)(m0l)      * DIMC + k, DIMC);
    v16h a1 = load_frag(lA + (size_t)(m0l + 16) * DIMC + k, DIMC);
    v16h b0 = load_frag(whT + (size_t)(n0)      * DIMC + k, DIMC);
    v16h b1 = load_frag(whT + (size_t)(n0 + 16) * DIMC + k, DIMC);
    acc[0][0] = wmma_f16(a0, b0, acc[0][0]);
    acc[0][1] = wmma_f16(a0, b1, acc[0][1]);
    acc[1][0] = wmma_f16(a1, b0, acc[1][0]);
    acc[1][1] = wmma_f16(a1, b1, acc[1][1]);
  }
  const int lane = threadIdx.x & 31;
  const int cn = lane & 15;
  const int rh = (lane >> 4) * 8;
#pragma unroll
  for (int mi = 0; mi < 2; ++mi)
#pragma unroll
    for (int ni = 0; ni < 2; ++ni) {
      const int ncol = n0 + ni * 16 + cn;
      const float bias = bh[ncol];
#pragma unroll
      for (int r = 0; r < 8; ++r) {
        const int mrow = mblk + m0l + mi * 16 + rh + r;
        const float v = silu_f(acc[mi][ni][r] + bias);
        const int bidx = mrow >> 11;            // batch
        const int srow = mrow & (SS - 1);       // seq pos
        if (ncol < HIDC)
          vT[((size_t)bidx * HIDC + ncol) * SS + srow] = (_Float16)v;
        else
          gate[(size_t)mrow * HIDC + (ncol - HIDC)] = (_Float16)v;
      }
    }
}

// ---------------------------------------------------------------------------
// K2: Z = silu(nx @ W_qk + b_qk); q = Z*gamma0+beta0, k = Z*gamma1+beta1 (f16)
// Same TDM-staged A-panel structure; N=128 -> grid.y = 2.
// ---------------------------------------------------------------------------
__global__ __launch_bounds__(256) void k_gemm_qk(
    const _Float16* __restrict__ nx, const _Float16* __restrict__ wqkT,
    const float* __restrict__ bqk, const float* __restrict__ gamma,
    const float* __restrict__ beta, _Float16* __restrict__ qh,
    _Float16* __restrict__ kh) {
  extern __shared__ __align__(16) _Float16 lA[];     // 128 x 512 f16
  const int wave = threadIdx.x >> 5;
  const int wm = wave >> 1, wn = wave & 1;
  const int mblk = blockIdx.x * 128;
  const int m0l = wm * 32;
  const int n0 = blockIdx.y * 64 + wn * 32;

  if (threadIdx.x < 32) {
    tdm_load_2d((unsigned int)(uintptr_t)lA, nx + (size_t)mblk * DIMC,
                DIMC * 2 / 8, 128);
    __builtin_amdgcn_s_wait_tensorcnt(0);
  }
  __syncthreads();

  v8f acc[2][2] = {};
  for (int k = 0; k < DIMC; k += 32) {
    v16h a0 = load_frag(lA + (size_t)(m0l)      * DIMC + k, DIMC);
    v16h a1 = load_frag(lA + (size_t)(m0l + 16) * DIMC + k, DIMC);
    v16h b0 = load_frag(wqkT + (size_t)(n0)      * DIMC + k, DIMC);
    v16h b1 = load_frag(wqkT + (size_t)(n0 + 16) * DIMC + k, DIMC);
    acc[0][0] = wmma_f16(a0, b0, acc[0][0]);
    acc[0][1] = wmma_f16(a0, b1, acc[0][1]);
    acc[1][0] = wmma_f16(a1, b0, acc[1][0]);
    acc[1][1] = wmma_f16(a1, b1, acc[1][1]);
  }
  const int lane = threadIdx.x & 31;
  const int cn = lane & 15;
  const int rh = (lane >> 4) * 8;
#pragma unroll
  for (int mi = 0; mi < 2; ++mi)
#pragma unroll
    for (int ni = 0; ni < 2; ++ni) {
      const int ncol = n0 + ni * 16 + cn;
      const float bias = bqk[ncol];
      const float g0 = gamma[ncol],        b0v = beta[ncol];
      const float g1 = gamma[QKD + ncol],  b1v = beta[QKD + ncol];
#pragma unroll
      for (int r = 0; r < 8; ++r) {
        const int mrow = mblk + m0l + mi * 16 + rh + r;
        const float z = silu_f(acc[mi][ni][r] + bias);
        qh[(size_t)mrow * QKD + ncol] = (_Float16)(z * g0 + b0v);
        kh[(size_t)mrow * QKD + ncol] = (_Float16)(z * g1 + b1v);
      }
    }
}

// ---------------------------------------------------------------------------
// K3: fused attention. One WG per (64-query strip, batch).
// Phase 1: Asim[64][2048] = relu(q@k^T / S)^2 into 256 KB LDS (f16).
// Phase 2: V = Asim @ v (B operand from vT), epilogue *gate -> Vg (f16).
// ---------------------------------------------------------------------------
__global__ __launch_bounds__(256) void k_attention(
    const _Float16* __restrict__ qh, const _Float16* __restrict__ kh,
    const _Float16* __restrict__ vT, const _Float16* __restrict__ gate,
    _Float16* __restrict__ Vg) {
  extern __shared__ __align__(16) _Float16 Asim[];   // 64 x 2048 f16 = 256 KB
  const int w    = threadIdx.x >> 5;
  const int lane = threadIdx.x & 31;
  const int cn   = lane & 15;
  const int rh   = (lane >> 4) * 8;
  const int i0   = blockIdx.x * 64;     // query strip base (within batch)
  const int bidx = blockIdx.y;

  // ---- Phase 1: sim strip. Wave w owns j-columns [w*256, w*256+256). ----
  const _Float16* qb = qh + ((size_t)bidx * SS + i0) * QKD;
  v16h qf[4][4];
#pragma unroll
  for (int mt = 0; mt < 4; ++mt)
#pragma unroll
    for (int ks = 0; ks < 4; ++ks)
      qf[mt][ks] = load_frag(qb + (size_t)(mt * 16) * QKD + ks * 32, QKD);

  const float inv_s = 1.0f / (float)SS;
  for (int nt = 0; nt < 16; ++nt) {
    const int j0 = w * 256 + nt * 16;
    v8f acc[4] = {};
#pragma unroll
    for (int ks = 0; ks < 4; ++ks) {
      v16h kf = load_frag(kh + ((size_t)bidx * SS + j0) * QKD + ks * 32, QKD);
#pragma unroll
      for (int mt = 0; mt < 4; ++mt) acc[mt] = wmma_f16(qf[mt][ks], kf, acc[mt]);
    }
#pragma unroll
    for (int mt = 0; mt < 4; ++mt)
#pragma unroll
      for (int r = 0; r < 8; ++r) {
        float v = acc[mt][r] * inv_s;
        v = (v > 0.0f) ? v * v : 0.0f;           // relu^2
        Asim[(size_t)(mt * 16 + rh + r) * SS + j0 + cn] = (_Float16)v;
      }
  }
  __syncthreads();

  // ---- Phase 2: V = Asim @ v. Waves: 2(m) x 4(d), two 128-wide d passes. --
  const int wm = w & 1, wn = w >> 1;
#pragma unroll 1
  for (int hd = 0; hd < 2; ++hd) {
    const int d0 = wn * 256 + hd * 128;
    v8f acc[2][8] = {};
    for (int j = 0; j < SS; j += 32) {
      __builtin_prefetch(vT + ((size_t)bidx * HIDC + d0) * SS + j + 256, 0, 1);
      v16h a0 = load_frag(Asim + (size_t)(wm * 32)      * SS + j, SS);
      v16h a1 = load_frag(Asim + (size_t)(wm * 32 + 16) * SS + j, SS);
#pragma unroll
      for (int nt = 0; nt < 8; ++nt) {
        v16h bf = load_frag(vT + ((size_t)bidx * HIDC + d0 + nt * 16) * SS + j, SS);
        acc[0][nt] = wmma_f16(a0, bf, acc[0][nt]);
        acc[1][nt] = wmma_f16(a1, bf, acc[1][nt]);
      }
    }
#pragma unroll
    for (int mi = 0; mi < 2; ++mi)
#pragma unroll
      for (int nt = 0; nt < 8; ++nt)
#pragma unroll
        for (int r = 0; r < 8; ++r) {
          const int srow = i0 + wm * 32 + mi * 16 + rh + r;
          const int dcol = d0 + nt * 16 + cn;
          const size_t idx = ((size_t)bidx * SS + srow) * HIDC + dcol;
          const float g = (float)gate[idx];
          Vg[idx] = (_Float16)(acc[mi][nt][r] * g);
        }
  }
}

// ---------------------------------------------------------------------------
// K4: out = Vg @ W_out + b_out + x   (BSR x 512, K = 1024) -> f32
// A-panel 128x1024 f16 = 256 KB staged via TDM.
// ---------------------------------------------------------------------------
__global__ __launch_bounds__(256) void k_gemm_out(
    const _Float16* __restrict__ Vg, const _Float16* __restrict__ woutT,
    const float* __restrict__ bout, const float* __restrict__ x,
    float* __restrict__ out) {
  extern __shared__ __align__(16) _Float16 lA[];     // 128 x 1024 f16
  const int wave = threadIdx.x >> 5;
  const int wm = wave >> 1, wn = wave & 1;
  const int mblk = blockIdx.x * 128;
  const int m0l = wm * 32;
  const int n0 = blockIdx.y * 64 + wn * 32;

  if (threadIdx.x < 32) {
    tdm_load_2d((unsigned int)(uintptr_t)lA, Vg + (size_t)mblk * HIDC,
                HIDC * 2 / 8, 128);
    __builtin_amdgcn_s_wait_tensorcnt(0);
  }
  __syncthreads();

  v8f acc[2][2] = {};
  for (int k = 0; k < HIDC; k += 32) {
    __builtin_prefetch(woutT + (size_t)n0 * HIDC + k + 128, 0, 1);
    v16h a0 = load_frag(lA + (size_t)(m0l)      * HIDC + k, HIDC);
    v16h a1 = load_frag(lA + (size_t)(m0l + 16) * HIDC + k, HIDC);
    v16h b0 = load_frag(woutT + (size_t)(n0)      * HIDC + k, HIDC);
    v16h b1 = load_frag(woutT + (size_t)(n0 + 16) * HIDC + k, HIDC);
    acc[0][0] = wmma_f16(a0, b0, acc[0][0]);
    acc[0][1] = wmma_f16(a0, b1, acc[0][1]);
    acc[1][0] = wmma_f16(a1, b0, acc[1][0]);
    acc[1][1] = wmma_f16(a1, b1, acc[1][1]);
  }
  const int lane = threadIdx.x & 31;
  const int cn = lane & 15;
  const int rh = (lane >> 4) * 8;
#pragma unroll
  for (int mi = 0; mi < 2; ++mi)
#pragma unroll
    for (int ni = 0; ni < 2; ++ni) {
      const int ncol = n0 + ni * 16 + cn;
      const float bias = bout[ncol];
#pragma unroll
      for (int r = 0; r < 8; ++r) {
        const int mrow = mblk + m0l + mi * 16 + rh + r;
        const size_t idx = (size_t)mrow * DIMC + ncol;
        out[idx] = acc[mi][ni][r] + bias + x[idx];
      }
    }
}

// ---------------------------------------------------------------------------
// Host-side orchestration
// ---------------------------------------------------------------------------
extern "C" void kernel_launch(void* const* d_in, const int* in_sizes, int n_in,
                              void* d_out, int out_size, void* d_ws, size_t ws_size,
                              hipStream_t stream) {
  (void)in_sizes; (void)n_in; (void)out_size; (void)ws_size;
  const float* x        = (const float*)d_in[0];
  const float* ln_g     = (const float*)d_in[1];
  const float* ln_b     = (const float*)d_in[2];
  const float* W_hidden = (const float*)d_in[3];
  const float* b_hidden = (const float*)d_in[4];
  const float* W_qk     = (const float*)d_in[5];
  const float* b_qk     = (const float*)d_in[6];
  const float* gamma    = (const float*)d_in[7];
  const float* beta     = (const float*)d_in[8];
  const float* W_out    = (const float*)d_in[9];
  const float* b_out    = (const float*)d_in[10];
  float* out = (float*)d_out;

  // Workspace layout (f16 buffers, 256B aligned slices)
  char* ws = (char*)d_ws;
  size_t off = 0;
  auto take = [&](size_t bytes) {
    char* p = ws + off;
    off += (bytes + 255) & ~(size_t)255;
    return p;
  };
  _Float16* nx    = (_Float16*)take((size_t)BSR * DIMC * 2);        // 16 MB
  _Float16* whT   = (_Float16*)take((size_t)2 * HIDC * DIMC * 2);   // 2 MB
  _Float16* wqkT  = (_Float16*)take((size_t)QKD * DIMC * 2);        // 128 KB
  _Float16* woutT = (_Float16*)take((size_t)DIMC * HIDC * 2);       // 1 MB
  _Float16* vT    = (_Float16*)take((size_t)BB * HIDC * SS * 2);    // 32 MB
  _Float16* gateb = (_Float16*)take((size_t)BSR * HIDC * 2);        // 32 MB
  _Float16* qh    = (_Float16*)take((size_t)BSR * QKD * 2);         // 4 MB
  _Float16* kh    = (_Float16*)take((size_t)BSR * QKD * 2);         // 4 MB
  _Float16* Vgb   = (_Float16*)take((size_t)BSR * HIDC * 2);        // 32 MB

  // 0: layernorm (one wave per row; 8 rows per 256-thread block)
  k_layernorm<<<BSR / 8, 256, 0, stream>>>(x, ln_g, ln_b, nx);

  // prep: weight transposes to f16 B^T layout
  k_transpose_w<<<(DIMC * 2 * HIDC + 255) / 256, 256, 0, stream>>>(W_hidden, whT, DIMC, 2 * HIDC);
  k_transpose_w<<<(DIMC * QKD + 255) / 256, 256, 0, stream>>>(W_qk, wqkT, DIMC, QKD);
  k_transpose_w<<<(HIDC * DIMC + 255) / 256, 256, 0, stream>>>(W_out, woutT, HIDC, DIMC);

  // 1: hidden GEMM + SiLU -> vT, gate  (TDM-staged A panel, 128 KB LDS)
  k_gemm_hidden<<<dim3(BSR / 128, (2 * HIDC) / 64), 256,
                  (size_t)128 * DIMC * 2, stream>>>(nx, whT, b_hidden, vT, gateb);

  // 2: qk GEMM + SiLU + affine -> q, k  (TDM-staged A panel)
  k_gemm_qk<<<dim3(BSR / 128, QKD / 64), 256,
              (size_t)128 * DIMC * 2, stream>>>(nx, wqkT, b_qk, gamma, beta, qh, kh);

  // 3: fused attention (256 KB dynamic LDS strip per WGP)
  k_attention<<<dim3(SS / 64, BB), 256, (size_t)64 * SS * 2, stream>>>(
      qh, kh, vT, gateb, Vgb);

  // 4: output GEMM + bias + residual  (TDM-staged A panel, 256 KB LDS)
  k_gemm_out<<<dim3(BSR / 128, DIMC / 64), 256,
               (size_t)128 * HIDC * 2, stream>>>(Vgb, woutT, b_out, x, out);
}
